// TAM_19327352832611
// MI455X (gfx1250) — matrix operand order repaired and checked by reading, not compiled, not run on hardware
//
#include <hip/hip_runtime.h>
#include <hip/hip_bf16.h>
#include <math.h>

// ---------------------------------------------------------------------------
// Problem constants (from reference)
// ---------------------------------------------------------------------------
#define NS   8
#define ALPHA 2
#define KK   3
#define C_   512
#define CB_  128
#define NB_  8
#define H_   32
#define W_   32
#define HW   (H_ * W_)          // 1024
#define NROW (NB_ * NS * C_)    // 32768 (n,c) rows
#define EPSV 1e-5f

typedef __attribute__((ext_vector_type(16))) _Float16 v16h;
typedef __attribute__((ext_vector_type(8)))  _Float16 v8h;
typedef __attribute__((ext_vector_type(8)))  float    v8f;
typedef __attribute__((ext_vector_type(4)))  float    v4f;

// ---------------------------------------------------------------------------
// WMMA fragment loaders (layouts per CDNA5 ISA 7.12.2, wave32)
// A (16x32 f16): lane<16 -> row=lane, halfs j0..7 = K0..7, j8..15 = K16..23
//                lane>=16 -> row=lane-16, j0..7 = K8..15, j8..15 = K24..31
// B (32x16 f16), stored transposed [n][k]:
//                lane<16 -> col=lane, j0..7 = K0..7, j8..15 = K16..23 ; lane>=16 -> K8..15, K24..31
// C/D (16x16 f32): reg r -> M = r + (lane<16?0:8), N = lane&15
// ---------------------------------------------------------------------------
__device__ inline v16h load_fragA(const _Float16* base, int row, int rowstride,
                                  int kbase, int lane) {
  const _Float16* p = base + (size_t)row * rowstride + kbase;
  int off0 = (lane < 16) ? 0 : 8;
  int off1 = (lane < 16) ? 16 : 24;
  v8h lo = *(const v8h*)(p + off0);
  v8h hi = *(const v8h*)(p + off1);
  v16h a;
#pragma unroll
  for (int i = 0; i < 8; ++i) { a[i] = lo[i]; a[i + 8] = hi[i]; }
  return a;
}

__device__ inline v16h load_fragB(const _Float16* baseT, int col, int rowstride,
                                  int kbase, int lane) {
  const _Float16* p = baseT + (size_t)col * rowstride + kbase + ((lane < 16) ? 0 : 8);
  v8h lo = *(const v8h*)(p);
  v8h hi = *(const v8h*)(p + 16);
  v16h b;
#pragma unroll
  for (int i = 0; i < 8; ++i) { b[i] = lo[i]; b[i + 8] = hi[i]; }
  return b;
}

// ---------------------------------------------------------------------------
// Kernel 1: theta[nb][c][s] = mean_{h,w} x[nb*NS+s][c][h][w]
// One wave (32 lanes) reduces one (n,c) row of 1024 floats.
// ---------------------------------------------------------------------------
__global__ void tam_theta_kernel(const float* __restrict__ x,
                                 float* __restrict__ thetaF) {
  int wave = threadIdx.x >> 5;
  int lane = threadIdx.x & 31;
  int row  = blockIdx.x * 8 + wave;            // row = n*C_ + c, n in [0,64)
  const v4f* p = (const v4f*)(x + (size_t)row * HW);
  float s = 0.0f;
#pragma unroll
  for (int i = 0; i < 8; ++i) {
    v4f v = p[lane + 32 * i];
    s += v.x + v.y + v.z + v.w;
  }
#pragma unroll
  for (int off = 16; off > 0; off >>= 1) s += __shfl_xor(s, off, 32);
  if (lane == 0) {
    int n  = row >> 9;        // / C_
    int c  = row & (C_ - 1);
    int nb = n >> 3;
    int sg = n & 7;
    thetaF[(((nb << 9) + c) << 3) + sg] = s * (1.0f / (float)HW);
  }
}

// ---------------------------------------------------------------------------
// Kernel 2: prep — f16 conversions + padded im2col of theta for WMMA
//   Ah [cb][c*3+k]  = (f16) l_w1
//   A2h[c][cb]      = (f16) l_w2
//   Bt [n][c*3+k]   = (f16) theta_pad[nb][c][s+k-1], n = nb*8+s
// ---------------------------------------------------------------------------
__global__ void tam_prep_kernel(const float* __restrict__ l_w1,
                                const float* __restrict__ l_w2,
                                const float* __restrict__ thetaF,
                                _Float16* __restrict__ Ah,
                                _Float16* __restrict__ A2h,
                                _Float16* __restrict__ Bt) {
  int idx = blockIdx.x * 256 + threadIdx.x;    // up to 196608
  if (idx < CB_ * C_ * KK)  Ah[idx]  = (_Float16)l_w1[idx];
  if (idx < C_ * CB_)       A2h[idx] = (_Float16)l_w2[idx];
  if (idx < NB_ * NS * C_ * KK) {
    int n = idx / (C_ * KK);
    int r = idx - n * (C_ * KK);
    int c = r / KK;
    int k = r - c * KK;
    int nb = n >> 3, sg = n & 7;
    int sp = sg + k - 1;
    float v = (sp >= 0 && sp < NS) ? thetaF[(((nb << 9) + c) << 3) + sp] : 0.0f;
    Bt[idx] = (_Float16)v;
  }
}

// ---------------------------------------------------------------------------
// Kernel 3: G branch — per-(nb,c) tiny MLP + softmax -> kernel weights (K=3)
// ---------------------------------------------------------------------------
__global__ void tam_gbranch_kernel(const float* __restrict__ thetaF,
                                   const float* __restrict__ g_w1,
                                   const float* __restrict__ g_gamma,
                                   const float* __restrict__ g_beta,
                                   const float* __restrict__ g_mean,
                                   const float* __restrict__ g_var,
                                   const float* __restrict__ g_w2,
                                   float* __restrict__ kernelW) {
  int t = blockIdx.x * 256 + threadIdx.x;      // 0..4095 = nb*C_ + c
  float th[NS];
#pragma unroll
  for (int i = 0; i < NS; ++i) th[i] = thetaF[t * NS + i];
  float z[KK] = {0.f, 0.f, 0.f};
#pragma unroll
  for (int j = 0; j < NS * ALPHA; ++j) {
    float acc = 0.f;
#pragma unroll
    for (int i = 0; i < NS; ++i) acc += g_w1[j * NS + i] * th[i];
    acc = (acc - g_mean[j]) * rsqrtf(g_var[j] + EPSV) * g_gamma[j] + g_beta[j];
    acc = fmaxf(acc, 0.f);
#pragma unroll
    for (int k = 0; k < KK; ++k) z[k] += g_w2[k * (NS * ALPHA) + j] * acc;
  }
  float m = fmaxf(z[0], fmaxf(z[1], z[2]));
  float e0 = expf(z[0] - m), e1 = expf(z[1] - m), e2 = expf(z[2] - m);
  float inv = 1.0f / (e0 + e1 + e2);
  kernelW[t * KK + 0] = e0 * inv;
  kernelW[t * KK + 1] = e1 * inv;
  kernelW[t * KK + 2] = e2 * inv;
}

// ---------------------------------------------------------------------------
// Kernel 4: L branch conv1 as WMMA GEMM: U = relu(BN(Ah[128x1536] @ B[1536x64]))
// 32 wave-tiles (8 M-tiles x 4 N-tiles), K = 48 steps of 32.
// Output Uh stored transposed [n][cb] (f16) for the next GEMM.
// ---------------------------------------------------------------------------
__global__ void tam_lconv1_wmma(const _Float16* __restrict__ Ah,
                                const _Float16* __restrict__ Bt,
                                const float* __restrict__ l_gamma,
                                const float* __restrict__ l_beta,
                                const float* __restrict__ l_mean,
                                const float* __restrict__ l_var,
                                _Float16* __restrict__ Uh) {
  int lane = threadIdx.x & 31;
  int gw   = blockIdx.x * 8 + (threadIdx.x >> 5);  // 0..31
  int mt   = gw >> 2;                              // 0..7
  int nt   = gw & 3;                               // 0..3
  v8f acc = {};
  for (int kk = 0; kk < 48; ++kk) {
    int kbase = kk * 32;
    v16h a = load_fragA(Ah, mt * 16 + (lane & 15), C_ * KK, kbase, lane);
    v16h b = load_fragB(Bt, nt * 16 + (lane & 15), C_ * KK, kbase, lane);
    acc = __builtin_amdgcn_wmma_f32_16x16x32_f16(false, a, false, b,
                                                 (short)0, acc, false, false);
  }
  int mAdd = (lane < 16) ? 0 : 8;
  int ncol = nt * 16 + (lane & 15);
#pragma unroll
  for (int r = 0; r < 8; ++r) {
    int cb = mt * 16 + r + mAdd;
    float v = acc[r];
    v = (v - l_mean[cb]) * rsqrtf(l_var[cb] + EPSV) * l_gamma[cb] + l_beta[cb];
    v = fmaxf(v, 0.f);
    Uh[ncol * CB_ + cb] = (_Float16)v;
  }
}

// ---------------------------------------------------------------------------
// Kernel 5: L branch conv2 as WMMA GEMM: gate = sigmoid(A2h[512x128] @ U[128x64])
// 128 wave-tiles (32 M-tiles x 4 N-tiles), K = 4 steps of 32.
// ---------------------------------------------------------------------------
__global__ void tam_lconv2_wmma(const _Float16* __restrict__ A2h,
                                const _Float16* __restrict__ Uh,
                                float* __restrict__ gateF) {
  int lane = threadIdx.x & 31;
  int gw   = blockIdx.x * 8 + (threadIdx.x >> 5);  // 0..127
  int mt   = gw >> 2;                              // 0..31
  int nt   = gw & 3;                               // 0..3
  v8f acc = {};
#pragma unroll
  for (int kk = 0; kk < 4; ++kk) {
    int kbase = kk * 32;
    v16h a = load_fragA(A2h, mt * 16 + (lane & 15), CB_, kbase, lane);
    v16h b = load_fragB(Uh,  nt * 16 + (lane & 15), CB_, kbase, lane);
    acc = __builtin_amdgcn_wmma_f32_16x16x32_f16(false, a, false, b,
                                                 (short)0, acc, false, false);
  }
  int mAdd = (lane < 16) ? 0 : 8;
  int n    = nt * 16 + (lane & 15);
  int nb   = n >> 3, sg = n & 7;
#pragma unroll
  for (int r = 0; r < 8; ++r) {
    int c = mt * 16 + r + mAdd;
    float v = acc[r];
    gateF[((((nb << 9) + c)) << 3) + sg] = 1.0f / (1.0f + expf(-v));
  }
}

// ---------------------------------------------------------------------------
// Kernel 6: fused gating + dynamic 3-tap temporal conv (bandwidth-bound pass).
// Block = one (nb, c): loads all 8 segments' float4 columns once (NT/last-use),
// produces all 8 temporal outputs, NT-stores result.
//   y[s] = sum_k kernelW[k] * gate[s+k-1] * x[s+k-1]   (zero-padded)
// ---------------------------------------------------------------------------
__global__ void tam_final_kernel(const float* __restrict__ x,
                                 const float* __restrict__ gateF,
                                 const float* __restrict__ kernelW,
                                 float* __restrict__ out) {
  int b  = blockIdx.x;            // 0..4095 = nb*C_ + c
  int nb = b >> 9;
  int c  = b & (C_ - 1);

  float g[NS];
#pragma unroll
  for (int s = 0; s < NS; ++s) g[s] = gateF[b * NS + s];
  float kw0 = kernelW[b * KK + 0];
  float kw1 = kernelW[b * KK + 1];
  float kw2 = kernelW[b * KK + 2];

  // combined weights: w[s][k] = kw[k] * gate[s+k-1] (0 out of range)
  float wA[NS], wB[NS], wC[NS];
#pragma unroll
  for (int s = 0; s < NS; ++s) {
    wA[s] = (s - 1 >= 0) ? kw0 * g[s - 1] : 0.f;
    wB[s] = kw1 * g[s];
    wC[s] = (s + 1 < NS) ? kw2 * g[s + 1] : 0.f;
  }

  v4f xs[NS];
#pragma unroll
  for (int s = 0; s < NS; ++s) {
    const v4f* p = (const v4f*)(x + ((size_t)((nb * NS + s) * C_ + c)) * HW);
    xs[s] = __builtin_nontemporal_load(p + threadIdx.x);   // last use of x
  }
#pragma unroll
  for (int s = 0; s < NS; ++s) {
    v4f y = wB[s] * xs[s];
    if (s > 0)      y += wA[s] * xs[s - 1];
    if (s < NS - 1) y += wC[s] * xs[s + 1];
    v4f* q = (v4f*)(out + ((size_t)((nb * NS + s) * C_ + c)) * HW);
    __builtin_nontemporal_store(y, q + threadIdx.x);       // streaming output
  }
}

// ---------------------------------------------------------------------------
// Launch
// ---------------------------------------------------------------------------
extern "C" void kernel_launch(void* const* d_in, const int* in_sizes, int n_in,
                              void* d_out, int out_size, void* d_ws, size_t ws_size,
                              hipStream_t stream) {
  const float* x        = (const float*)d_in[0];
  const float* g_w1     = (const float*)d_in[1];
  const float* g_gamma  = (const float*)d_in[2];
  const float* g_beta   = (const float*)d_in[3];
  const float* g_mean   = (const float*)d_in[4];
  const float* g_var    = (const float*)d_in[5];
  const float* g_w2     = (const float*)d_in[6];
  const float* l_w1     = (const float*)d_in[7];
  const float* l_gamma  = (const float*)d_in[8];
  const float* l_beta   = (const float*)d_in[9];
  const float* l_mean   = (const float*)d_in[10];
  const float* l_var    = (const float*)d_in[11];
  const float* l_w2     = (const float*)d_in[12];
  float* out = (float*)d_out;

  // workspace partition (256B aligned)
  char* ws = (char*)d_ws;
  size_t off = 0;
  auto wsalloc = [&](size_t bytes) -> void* {
    void* p = ws + off;
    off = (off + bytes + 255) & ~(size_t)255;
    return p;
  };
  float*    thetaF  = (float*)   wsalloc((size_t)NB_ * C_ * NS * 4);        // 128KB
  float*    kernelW = (float*)   wsalloc((size_t)NB_ * C_ * KK * 4);        // 48KB
  float*    gateF   = (float*)   wsalloc((size_t)NB_ * C_ * NS * 4);        // 128KB
  _Float16* Ah      = (_Float16*)wsalloc((size_t)CB_ * C_ * KK * 2);        // 384KB
  _Float16* A2h     = (_Float16*)wsalloc((size_t)C_ * CB_ * 2);             // 128KB
  _Float16* Bt      = (_Float16*)wsalloc((size_t)NB_ * NS * C_ * KK * 2);   // 192KB
  _Float16* Uh      = (_Float16*)wsalloc((size_t)NB_ * NS * CB_ * 2);       // 16KB

  // 1) spatial pooling -> theta
  tam_theta_kernel<<<NROW / 8, 256, 0, stream>>>(x, thetaF);
  // 2) f16 conversions + padded im2col of theta
  tam_prep_kernel<<<(CB_ * C_ * KK + 255) / 256, 256, 0, stream>>>(
      l_w1, l_w2, thetaF, Ah, A2h, Bt);
  // 3) G branch -> adaptive kernel weights
  tam_gbranch_kernel<<<(NB_ * C_) / 256, 256, 0, stream>>>(
      thetaF, g_w1, g_gamma, g_beta, g_mean, g_var, g_w2, kernelW);
  // 4) L branch conv1 (WMMA GEMM 128x64x1536) + BN + ReLU
  tam_lconv1_wmma<<<4, 256, 0, stream>>>(Ah, Bt, l_gamma, l_beta, l_mean, l_var, Uh);
  // 5) L branch conv2 (WMMA GEMM 512x64x128) + sigmoid -> gate
  tam_lconv2_wmma<<<16, 256, 0, stream>>>(A2h, Uh, gateF);
  // 6) fused gating + dynamic temporal conv (streaming, NT loads/stores)
  tam_final_kernel<<<NB_ * C_, 256, 0, stream>>>(x, gateF, kernelW, out);
}
